// SASA_6811818131621
// MI455X (gfx1250) — matrix operand order repaired and checked by reading, not compiled, not run on hardware
//
#include <hip/hip_runtime.h>
#include <stdint.h>

typedef __attribute__((ext_vector_type(16))) _Float16 v16h;
typedef __attribute__((ext_vector_type(8)))  _Float16 v8h;
typedef __attribute__((ext_vector_type(8)))  float    v8f;
typedef __attribute__((ext_vector_type(4)))  float    v4f;
typedef __attribute__((ext_vector_type(4)))  unsigned int v4u;
typedef __attribute__((ext_vector_type(8)))  int      v8i;
typedef __attribute__((ext_vector_type(4)))  int      v4i;

#define BB    2
#define CC    256
#define HH    48
#define WW    48
#define HWSZ  (HH * WW)          // 2304
#define HEADS 4
#define DHD   64
#define OCC   256
#define HP    52                 // padded height (48 + 2 + 2)
#define WP    52
#define QSCALE 0.125f            // 64^-0.5

// Fragment layouts (wave32, 16-bit A/B, f32 C/D), half = lane>>4, ml = lane&15:
//  A (16x32): lane row = ml; elements [0..7]  = K (8*half)+0..7,
//                           elements [8..15] = K 16+(8*half)+0..7
//  B (32x16): lane col = ml; elements = same K order as A (sasa_attn: K (16*half)+0..15 for both)
//  C (16x16): VGPR r -> row (r + 8*half), col = ml

// ===========================================================================
// Kernel 1: q projection.  Qh[b][head][pix][dh] = f16( SCALE * Wq @ qmap )
// ===========================================================================
__global__ void sasa_proj_q(const float* __restrict__ qmap,
                            const float* __restrict__ Wq,
                            _Float16* __restrict__ Qh) {
    const int lane = threadIdx.x, half = lane >> 4, ml = lane & 15;
    const int b    = blockIdx.x / (HWSZ / 16);
    const int pix0 = (blockIdx.x % (HWSZ / 16)) * 16;
    const int oc0  = blockIdx.y * 16;

    const float* __restrict__ Arow = Wq + (oc0 + ml) * CC + half * 8;
    const float* __restrict__ Bcol = qmap + b * CC * HWSZ + half * 8 * HWSZ
                                     + pix0 + ml;
    v8f acc = {};
#pragma unroll
    for (int kc = 0; kc < CC; kc += 32) {
        v16h a, bf;
        const v4f a0 = *(const v4f*)(Arow + kc);
        const v4f a1 = *(const v4f*)(Arow + kc + 4);
        const v4f a2 = *(const v4f*)(Arow + kc + 16);
        const v4f a3 = *(const v4f*)(Arow + kc + 20);
#pragma unroll
        for (int e = 0; e < 4; ++e) {
            a[e]      = (_Float16)a0[e];
            a[e + 4]  = (_Float16)a1[e];
            a[e + 8]  = (_Float16)a2[e];
            a[e + 12] = (_Float16)a3[e];
        }
#pragma unroll
        for (int e = 0; e < 16; ++e)
            bf[e] = (_Float16)Bcol[(kc + e + (e & 8)) * HWSZ];
        acc = __builtin_amdgcn_wmma_f32_16x16x32_f16(false, a, false, bf,
                                                     (short)0, acc, false, false);
    }
    // 8 C-rows = 8 consecutive dh -> one b128 store
    const int head = oc0 >> 6;
    const int dh0  = (oc0 & 63) + 8 * half;
    v8h pk;
#pragma unroll
    for (int r = 0; r < 8; ++r) pk[r] = (_Float16)(acc[r] * QSCALE);
    *(v8h*)(Qh + ((b * HEADS + head) * HWSZ + pix0 + ml) * DHD + dh0) = pk;
}

// ===========================================================================
// Kernel 2: kv projection into zero-padded buffer Kh[b][l][hp][wp][oc] (f16)
// ===========================================================================
__global__ void sasa_proj_kv(const float* __restrict__ kv1,
                             const float* __restrict__ kv2,
                             const float* __restrict__ Wkv,
                             _Float16* __restrict__ Kh) {
    const int lane = threadIdx.x, half = lane >> 4, ml = lane & 15;
    int t = blockIdx.x;
    const int wt = t & 3;  t >>= 2;
    const int hp = t % HP; t /= HP;
    const int l  = t & 1;
    const int b  = t >> 1;
    const int oc0 = blockIdx.y * 16;

    const float* __restrict__ X = (l == 0 ? kv1 : kv2) + b * CC * HWSZ;
    const int h  = hp - 2;
    const int wp = wt * 16 + ml;
    const int w  = wp - 2;
    const bool cv = (h >= 0) && (h < HH) && (w >= 0) && (w < WW);

    const float* __restrict__ Arow = Wkv + (oc0 + ml) * CC + half * 8;
    const float* __restrict__ Bcol = X + (cv ? h * WW + w : 0) + half * 8 * HWSZ;

    v8f acc = {};
#pragma unroll
    for (int kc = 0; kc < CC; kc += 32) {
        v16h a, bf;
        const v4f a0 = *(const v4f*)(Arow + kc);
        const v4f a1 = *(const v4f*)(Arow + kc + 4);
        const v4f a2 = *(const v4f*)(Arow + kc + 16);
        const v4f a3 = *(const v4f*)(Arow + kc + 20);
#pragma unroll
        for (int e = 0; e < 4; ++e) {
            a[e]      = (_Float16)a0[e];
            a[e + 4]  = (_Float16)a1[e];
            a[e + 8]  = (_Float16)a2[e];
            a[e + 12] = (_Float16)a3[e];
        }
#pragma unroll
        for (int e = 0; e < 16; ++e)
            bf[e] = cv ? (_Float16)Bcol[(kc + e + (e & 8)) * HWSZ] : (_Float16)0.f;
        acc = __builtin_amdgcn_wmma_f32_16x16x32_f16(false, a, false, bf,
                                                     (short)0, acc, false, false);
    }
    if (wp < WP) {
        v8h pk;
#pragma unroll
        for (int r = 0; r < 8; ++r) pk[r] = (_Float16)acc[r];
        const int dh0 = oc0 + 8 * half;   // oc0..oc0+15 are consecutive oc
        *(v8h*)(Kh + (((b * 2 + l) * HP + hp) * WP + wp) * OCC + dh0) = pk;
    }
}

// ===========================================================================
// Kernel 3: windowed attention. One wave per 4x4 query tile.
//   Key tile (2 x 8 x 8 keys x 64 dh, f16) is DMA'd into LDS by the
//   Tensor Data Mover (one 3D-tile descriptor per level), then
//   logits GEMM (16x128, K=64) -> mask -> softmax -> attn GEMM (16x64, K=128).
// ===========================================================================
__global__ void sasa_attn(const _Float16* __restrict__ Qh,
                          const _Float16* __restrict__ Kh,
                          float* __restrict__ out) {
    __shared__ _Float16 Klds[128 * 64];   // [key y = l*64+i*8+j][dh]
    __shared__ _Float16 Wlds[16 * 128];   // [pixel p][key y]

    const int lane = threadIdx.x, half = lane >> 4, ml = lane & 15;
    const int tw = blockIdx.x % (WW / 4), th = blockIdx.x / (WW / 4);
    const int head = blockIdx.y % HEADS,  b  = blockIdx.y / HEADS;
    const int h0 = th * 4, w0 = tw * 4;

    // ---- TDM: DMA the two 64(dh) x 8(j) x 8(i) tiles into LDS -------------
    const unsigned lds_base = (unsigned)(uintptr_t)(void*)Klds;
#pragma unroll
    for (int l = 0; l < 2; ++l) {
        const _Float16* gptr =
            Kh + ((((b * 2 + l) * HP + h0) * WP + w0) * OCC + head * DHD);
        const uint64_t ga = (uint64_t)(uintptr_t)gptr;
        const v4u g0 = { 1u,                                  // count = 1
                         lds_base + (unsigned)l * 8192u,      // lds_addr
                         (unsigned)(ga & 0xffffffffu),        // global_addr lo
                         ((unsigned)(ga >> 32) & 0x01ffffffu) // global_addr hi
                             | (2u << 30) };                  // type = 2
        const v8i g1 = { (int)(1u << 16),            // data_size = 2 bytes
                         (int)(64u << 16),           // tensor_dim0 = 64
                         (int)(8u << 16),            // tensor_dim1 = 8
                         (int)(64u << 16),           // tile_dim0 = 64
                         (int)(8u | (8u << 16)),     // tile_dim1=8, tile_dim2=8
                         OCC,                        // tensor_dim0_stride = 256
                         (int)((unsigned)(WP * OCC) << 16), // dim1_stride lo16
                         0 };                        // dim1_stride hi
        const v4i g2 = { 8, 0, 0, 0 };               // tensor_dim2 = 8
        const v4i g3 = { 0, 0, 0, 0 };
        const v8i g4 = { 0, 0, 0, 0, 0, 0, 0, 0 };   // unused VADDR4 slot
        __builtin_amdgcn_tensor_load_to_lds(g0, g1, g2, g3, g4, 0);
    }
    __builtin_amdgcn_s_wait_tensorcnt((short)0);
    __syncthreads();

    // ---- Q A-fragments (two contiguous 16B runs per chunk) ----------------
    const _Float16* __restrict__ qrow =
        Qh + ((b * HEADS + head) * HWSZ + (h0 + (ml >> 2)) * WW + w0 + (ml & 3))
                 * DHD + half * 16;
    v16h qa[2];
#pragma unroll
    for (int c = 0; c < 2; ++c) {
        *(v8h*)&qa[c]       = *(const v8h*)(qrow + c * 32);
        *((v8h*)&qa[c] + 1) = *(const v8h*)(qrow + c * 32 + 8);
    }

    // ---- logits: B-fragment is one contiguous 32B LDS read ----------------
    v8f lg[8];
#pragma unroll
    for (int yt = 0; yt < 8; ++yt) {
        v8f acc = {};
        const int y = yt * 16 + ml;
#pragma unroll
        for (int c = 0; c < 2; ++c) {
            const v16h bf = *(const v16h*)(Klds + y * 64 + c * 32 + half * 16);
            acc = __builtin_amdgcn_wmma_f32_16x16x32_f16(false, qa[c], false, bf,
                                                         (short)0, acc, false, false);
        }
        lg[yt] = acc;
    }

    // ---- mask (5x5 window) + softmax over 128 keys ------------------------
    float rmax[8], rsum[8];
#pragma unroll
    for (int r = 0; r < 8; ++r) rmax[r] = -3.0e38f;
#pragma unroll
    for (int yt = 0; yt < 8; ++yt) {
        const int y = yt * 16 + ml, ki = (y >> 3) & 7, kj = y & 7;
#pragma unroll
        for (int r = 0; r < 8; ++r) {
            const int px = r + 8 * half, pi = px >> 2, pj = px & 3;
            const bool ok = ((unsigned)(ki - pi) <= 4u) && ((unsigned)(kj - pj) <= 4u);
            const float v = ok ? lg[yt][r] : -3.0e38f;
            lg[yt][r] = v;
            rmax[r] = fmaxf(rmax[r], v);
        }
    }
#pragma unroll
    for (int off = 1; off < 16; off <<= 1)
#pragma unroll
        for (int r = 0; r < 8; ++r)
            rmax[r] = fmaxf(rmax[r], __shfl_xor(rmax[r], off, 32));
#pragma unroll
    for (int r = 0; r < 8; ++r) rsum[r] = 0.f;
#pragma unroll
    for (int yt = 0; yt < 8; ++yt)
#pragma unroll
        for (int r = 0; r < 8; ++r) {
            const float e = (lg[yt][r] > -1.0e38f) ? __expf(lg[yt][r] - rmax[r]) : 0.f;
            lg[yt][r] = e;
            rsum[r] += e;
        }
#pragma unroll
    for (int off = 1; off < 16; off <<= 1)
#pragma unroll
        for (int r = 0; r < 8; ++r)
            rsum[r] += __shfl_xor(rsum[r], off, 32);

    // ---- normalized weights -> LDS (transpose into A layout) --------------
#pragma unroll
    for (int yt = 0; yt < 8; ++yt)
#pragma unroll
        for (int r = 0; r < 8; ++r)
            Wlds[(r + 8 * half) * 128 + yt * 16 + ml] =
                (_Float16)(lg[yt][r] / rsum[r]);
    __syncthreads();

    // ---- attn = W (16x128) @ K (128x64) -----------------------------------
    v8f o[4] = {};
#pragma unroll
    for (int c = 0; c < 4; ++c) {
        v16h wa;
        const _Float16* wrow = Wlds + ml * 128 + c * 32 + half * 16;
        *(v8h*)&wa       = *(const v8h*)(wrow);
        *((v8h*)&wa + 1) = *(const v8h*)(wrow + 8);
#pragma unroll
        for (int dt = 0; dt < 4; ++dt) {
            const int d = dt * 16 + ml;
            v16h bf;
#pragma unroll
            for (int v = 0; v < 8; ++v) {
                const int k0 = c * 32 + half * 16 + 2 * v;   // key index pair
                bf[2 * v]     = Klds[k0 * 64 + d];
                bf[2 * v + 1] = Klds[(k0 + 1) * 64 + d];
            }
            o[dt] = __builtin_amdgcn_wmma_f32_16x16x32_f16(false, wa, false, bf,
                                                           (short)0, o[dt], false, false);
        }
    }

    // ---- store: rows 0-3 / 4-7 of each C-tile are 4 consecutive pixels ----
#pragma unroll
    for (int dt = 0; dt < 4; ++dt) {
        const int oc = head * DHD + dt * 16 + ml;
        float* o0 = out + (b * OCC + oc) * HWSZ + (h0 + 2 * half) * WW + w0;
        const v4f s0 = { o[dt][0], o[dt][1], o[dt][2], o[dt][3] };
        const v4f s1 = { o[dt][4], o[dt][5], o[dt][6], o[dt][7] };
        *(v4f*)o0        = s0;
        *(v4f*)(o0 + WW) = s1;
    }
}

// ===========================================================================
extern "C" void kernel_launch(void* const* d_in, const int* in_sizes, int n_in,
                              void* d_out, int out_size, void* d_ws, size_t ws_size,
                              hipStream_t stream) {
    const float* kvmap1 = (const float*)d_in[0];
    const float* qmap   = (const float*)d_in[1];
    const float* kvmap2 = (const float*)d_in[2];
    const float* Wq     = (const float*)d_in[3];
    const float* Wkv    = (const float*)d_in[4];
    float* out = (float*)d_out;

    _Float16* Qh = (_Float16*)d_ws;                                  // 2.36 MB
    size_t qbytes = (size_t)BB * HEADS * HWSZ * DHD * sizeof(_Float16);
    qbytes = (qbytes + 255) & ~(size_t)255;
    _Float16* Kh = (_Float16*)((char*)d_ws + qbytes);                // 5.54 MB

    sasa_proj_q <<<dim3(BB * (HWSZ / 16), OCC / 16), 32, 0, stream>>>(qmap, Wq, Qh);
    sasa_proj_kv<<<dim3(BB * 2 * HP * 4, OCC / 16), 32, 0, stream>>>(kvmap1, kvmap2,
                                                                     Wkv, Kh);
    sasa_attn   <<<dim3((HH / 4) * (WW / 4), BB * HEADS), 32, 0, stream>>>(Qh, Kh, out);
}
